// PointNet2_model_46583215292842
// MI455X (gfx1250) — compile-verified, matrix operand order checked by reading, not run on hardware
//
#include <hip/hip_runtime.h>
#include <hip/hip_bf16.h>

// ---------------------------------------------------------------------------
// PointNet++ forward for MI455X (gfx1250), wave32 + WMMA f16 16x16x32.
// GEMM uses CDNA5 async global->LDS copies (ASYNCcnt) with double-buffered
// LDS tiles; 128x128 block tile, 8 waves x (64x32) per wave = 8 WMMA/K-step.
// Linear bias is dropped (cancels exactly under training-mode batchnorm).
// ---------------------------------------------------------------------------

typedef __attribute__((ext_vector_type(16))) _Float16 v16h;
typedef __attribute__((ext_vector_type(8)))  float    v8f;

union F16Frag { v16h v; uint4 q[2]; };

constexpr int BQ = 8;              // B0 * NPART
constexpr int STATS_STRIDE = 4096; // per-channel stats stride
__device__ __forceinline__ float relu_(float v) { return v > 0.f ? v : 0.f; }

// Low 32 bits of a generic pointer to __shared__ = LDS byte offset
__device__ __forceinline__ unsigned lds_off(const void* p) {
    return (unsigned)(size_t)p;
}

// CDNA5 async copy: 16 bytes/lane, global -> LDS, tracked by ASYNCcnt.
__device__ __forceinline__ void async_load_b128(const void* gaddr, unsigned ldsaddr) {
    asm volatile("global_load_async_to_lds_b128 %0, %1, off"
                 :: "v"(ldsaddr), "v"(gaddr) : "memory");
}
__device__ __forceinline__ void wait_asynccnt0() {
    asm volatile("s_wait_asynccnt 0x0" ::: "memory");
}

// ---------------------------------------------------------------------------
// Farthest point sampling: one block per point cloud.
// ---------------------------------------------------------------------------
__global__ __launch_bounds__(256)
void fps_kernel(const float* __restrict__ xyz, int N, int S, int* __restrict__ idx)
{
    const int b = blockIdx.x;
    const float* X = xyz + (size_t)b * N * 3;
    __shared__ float dist[2048];
    __shared__ float rv[256];
    __shared__ int   ri[256];
    __shared__ int   cur;
    for (int i = threadIdx.x; i < N; i += 256) dist[i] = 1e10f;
    if (threadIdx.x == 0) cur = 0;
    __syncthreads();
    for (int it = 0; it < S; ++it) {
        const int f = cur;
        if (threadIdx.x == 0) idx[b * S + it] = f;
        const float cx = X[f*3], cy = X[f*3+1], cz = X[f*3+2];
        float bm = -1.f; int bi = 0;
        for (int i = threadIdx.x; i < N; i += 256) {
            const float dx = X[i*3] - cx, dy = X[i*3+1] - cy, dz = X[i*3+2] - cz;
            const float nd = fminf(dist[i], dx*dx + dy*dy + dz*dz);
            dist[i] = nd;
            if (nd > bm) { bm = nd; bi = i; }
        }
        rv[threadIdx.x] = bm; ri[threadIdx.x] = bi;
        __syncthreads();
        for (int st = 128; st > 0; st >>= 1) {
            if (threadIdx.x < st) {
                const float ov = rv[threadIdx.x + st];
                const int   oi = ri[threadIdx.x + st];
                if (ov > rv[threadIdx.x] || (ov == rv[threadIdx.x] && oi < ri[threadIdx.x])) {
                    rv[threadIdx.x] = ov; ri[threadIdx.x] = oi;
                }
            }
            __syncthreads();
        }
        if (threadIdx.x == 0) cur = ri[0];
        __syncthreads();
    }
}

__global__ void gather_newxyz(const float* __restrict__ xyz, const int* __restrict__ fidx,
                              float* __restrict__ nxyz, int N, int S, int Q)
{
    const int q = blockIdx.x * blockDim.x + threadIdx.x;
    if (q >= Q) return;
    const int b = q / S;
    const int gi = fidx[q];
    const float* p = xyz + ((size_t)b * N + gi) * 3;
    nxyz[q*3+0] = p[0]; nxyz[q*3+1] = p[1]; nxyz[q*3+2] = p[2];
}

// ---------------------------------------------------------------------------
// Ball query: one wave (32 lanes) per query point; ascending-index selection.
// ---------------------------------------------------------------------------
__global__ __launch_bounds__(256)
void ballquery_kernel(const float* __restrict__ xyz, const float* __restrict__ nxyz,
                      float r2, int N, int S, int Kc, int* __restrict__ idx, int Q)
{
    const int q = blockIdx.x * 8 + (threadIdx.x >> 5);
    const int lane = threadIdx.x & 31;
    if (q >= Q) return;
    const int b = q / S;
    const float* X = xyz + (size_t)b * N * 3;
    const float nx = nxyz[q*3], ny = nxyz[q*3+1], nz = nxyz[q*3+2];
    int cnt = 0, first = -1;
    for (int base = 0; base < N && cnt < Kc; base += 32) {
        const int p = base + lane;
        const float dx = X[p*3] - nx, dy = X[p*3+1] - ny, dz = X[p*3+2] - nz;
        unsigned long long mask = __ballot((dx*dx + dy*dy + dz*dz) <= r2);
        while (mask && cnt < Kc) {
            const int j = __builtin_ctzll(mask);
            if (first < 0) first = base + j;
            if (lane == 0) idx[(size_t)q * Kc + cnt] = base + j;
            ++cnt;
            mask &= mask - 1;
        }
    }
    if (first < 0) first = 0;
    if (lane == 0)
        for (int c = cnt; c < Kc; ++c) idx[(size_t)q * Kc + c] = first;
}

// ---------------------------------------------------------------------------
// Build f16 A-matrix rows: [xyz - new_xyz | features | zero pad]
// ---------------------------------------------------------------------------
__global__ void group_kernel(const float* __restrict__ xyz, const float* __restrict__ nxyz,
                             const float* __restrict__ pts, const int* __restrict__ idx,
                             _Float16* __restrict__ A, int N, int S, int Kc, int Cf, int Kpad)
{
    const int row = blockIdx.x;          // q*Kc + k
    const int q = row / Kc, k = row % Kc;
    const int b = q / S;
    const int gi = idx[(size_t)q * Kc + k];
    const float* xp = xyz  + ((size_t)b * N + gi) * 3;
    const float* np = nxyz + (size_t)q * 3;
    const float* pp = pts  + ((size_t)b * N + gi) * Cf;
    _Float16* ar = A + (size_t)row * Kpad;
    for (int c = threadIdx.x; c < Kpad; c += blockDim.x) {
        float v;
        if (c < 3)            v = xp[c] - np[c];
        else if (c < 3 + Cf)  v = pp[c - 3];
        else                  v = 0.f;
        ar[c] = (_Float16)v;
    }
}

__global__ void group_all_kernel(const float* __restrict__ xyz, const float* __restrict__ pts,
                                 _Float16* __restrict__ A, int N, int Cf, int Kpad)
{
    const int row = blockIdx.x;          // b*N + n
    const int b = row / N, n = row % N;
    const float* xp = xyz + ((size_t)b * N + n) * 3;
    const float* pp = pts + ((size_t)b * N + n) * Cf;
    _Float16* ar = A + (size_t)row * Kpad;
    for (int c = threadIdx.x; c < Kpad; c += blockDim.x) {
        float v;
        if (c < 3)            v = xp[c];
        else if (c < 3 + Cf)  v = pp[c - 3];
        else                  v = 0.f;
        ar[c] = (_Float16)v;
    }
}

__global__ void convert_w(const float* __restrict__ W, _Float16* __restrict__ Wh,
                          int Cout, int Cin, int Kpad)
{
    const int o = blockIdx.x;
    for (int c = threadIdx.x; c < Kpad; c += blockDim.x)
        Wh[(size_t)o * Kpad + c] = (_Float16)(c < Cin ? W[(size_t)o * Cin + c] : 0.f);
}

// ---------------------------------------------------------------------------
// WMMA GEMM: C[M,N] (f16) = A[M,K] (f16) x W[N,K]^T (f16), f32 accumulate.
// Block: 256 threads = 8 waves; 128x128 tile; wave (wr,wc) owns a 64x32
// subtile = 4x2 v_wmma_f32_16x16x32_f16 per K-step.  Tiles are staged into
// double-buffered LDS via global_load_async_to_lds_b128 (ASYNCcnt), so the
// next K-tile streams in while WMMAs consume the current one.
// Requirements: M % 128 == 0, N % 128 == 0, K % 32 == 0.
// ---------------------------------------------------------------------------
__global__ __launch_bounds__(256)
void gemm_f16_wmma(const _Float16* __restrict__ A, const _Float16* __restrict__ W,
                   _Float16* __restrict__ C, int M, int N, int K)
{
    __shared__ _Float16 As[2][128][32];
    __shared__ _Float16 Bs[2][128][32];
    const int t    = threadIdx.x;
    const int lane = t & 31;
    const int wave = t >> 5;
    const int wr = wave & 1;        // row half   (64 rows)
    const int wc = wave >> 1;       // col quarter (32 cols)
    const int m0 = blockIdx.y * 128;
    const int n0 = blockIdx.x * 128;

    // Per-thread staging slots: 128 rows x 4 x 16B chunks = 512 chunks per
    // matrix; 256 threads issue 2 async b128 per matrix per K-step.
    auto stage = [&](int buf, int k0) {
        #pragma unroll
        for (int i = 0; i < 2; ++i) {
            const int s = t * 2 + i;
            const int row = s >> 2, seg = s & 3;
            async_load_b128(A + (size_t)(m0 + row) * K + k0 + seg * 8,
                            lds_off(&As[buf][row][seg * 8]));
            async_load_b128(W + (size_t)(n0 + row) * K + k0 + seg * 8,
                            lds_off(&Bs[buf][row][seg * 8]));
        }
    };

    stage(0, 0);
    wait_asynccnt0();
    __syncthreads();

    v8f acc[4][2] = {};
    int cb = 0;
    for (int k0 = 0; k0 < K; k0 += 32) {
        if (k0 + 32 < K) stage(cb ^ 1, k0 + 32);   // overlap next tile

        // fragments from LDS in the ISA 16-bit WMMA lane layout:
        // lane L holds row (L&15), K chunks [kh*8,+8) and [16+kh*8,+8), kh=L>>4
        const int fr = lane & 15, kh = lane >> 4;
        F16Frag a[4], b[2];
        #pragma unroll
        for (int i = 0; i < 4; ++i) {
            const int ar = wr * 64 + i * 16 + fr;
            a[i].q[0] = *(const uint4*)&As[cb][ar][kh * 8];
            a[i].q[1] = *(const uint4*)&As[cb][ar][16 + kh * 8];
        }
        #pragma unroll
        for (int j = 0; j < 2; ++j) {
            const int br = wc * 32 + j * 16 + fr;
            b[j].q[0] = *(const uint4*)&Bs[cb][br][kh * 8];
            b[j].q[1] = *(const uint4*)&Bs[cb][br][16 + kh * 8];
        }
        #pragma unroll
        for (int i = 0; i < 4; ++i)
            #pragma unroll
            for (int j = 0; j < 2; ++j)
                acc[i][j] = __builtin_amdgcn_wmma_f32_16x16x32_f16(
                    false, a[i].v, false, b[j].v, (short)0, acc[i][j], false, false);

        wait_asynccnt0();   // next tile fully in LDS (this wave's part)
        __syncthreads();    // everyone's part visible; cur tile reads done
        cb ^= 1;
    }

    // store: C layout lane=(col | rowhalf), VGPR r -> row r + 8*rowhalf
    const int cn = lane & 15, rh = lane >> 4;
    #pragma unroll
    for (int i = 0; i < 4; ++i)
        #pragma unroll
        for (int j = 0; j < 2; ++j)
            #pragma unroll
            for (int r = 0; r < 8; ++r) {
                const int row = m0 + wr * 64 + i * 16 + r + 8 * rh;
                const int col = n0 + wc * 32 + j * 16 + cn;
                C[(size_t)row * N + col] = (_Float16)acc[i][j][r];
            }
}

// ---------------------------------------------------------------------------
// Batchnorm statistics (sum / sumsq per channel) with block reduce + atomics.
// ---------------------------------------------------------------------------
__global__ __launch_bounds__(256)
void bn_stats(const _Float16* __restrict__ H, float* __restrict__ stats,
              int M, int N, int rowsPerBlock)
{
    const int cl = threadIdx.x & 63;
    const int c  = blockIdx.x * 64 + cl;
    const int rg = threadIdx.x >> 6;   // 0..3
    const int r0 = blockIdx.y * rowsPerBlock;
    const int r1 = min(M, r0 + rowsPerBlock);
    float s = 0.f, s2 = 0.f;
    if (c < N)
        for (int r = r0 + rg; r < r1; r += 4) {
            const float v = (float)H[(size_t)r * N + c];
            s += v; s2 += v * v;
        }
    __shared__ float sh[2][4][64];
    sh[0][rg][cl] = s; sh[1][rg][cl] = s2;
    __syncthreads();
    if (rg == 0 && c < N) {
        const float ts = sh[0][0][cl] + sh[0][1][cl] + sh[0][2][cl] + sh[0][3][cl];
        const float t2 = sh[1][0][cl] + sh[1][1][cl] + sh[1][2][cl] + sh[1][3][cl];
        atomicAdd(&stats[c], ts);
        atomicAdd(&stats[STATS_STRIDE + c], t2);
    }
}

__global__ void bn_apply(_Float16* __restrict__ H, const float* __restrict__ stats,
                         const float* __restrict__ g, const float* __restrict__ be,
                         int M, int N)
{
    const size_t i = (size_t)blockIdx.x * blockDim.x + threadIdx.x;
    if (i >= (size_t)M * N) return;
    const int c = (int)(i % N);
    const float mean = stats[c] / M;
    const float var  = stats[STATS_STRIDE + c] / M - mean * mean;
    const float v = ((float)H[i] - mean) * rsqrtf(var + 1e-5f) * g[c] + be[c];
    H[i] = (_Float16)relu_(v);
}

__global__ void maxpool_kernel(const _Float16* __restrict__ H, float* __restrict__ out,
                               int Kc, int N)
{
    const int q = blockIdx.x;
    const _Float16* base = H + (size_t)q * Kc * N;
    for (int c = threadIdx.x; c < N; c += blockDim.x) {
        float m = -1e30f;
        for (int k = 0; k < Kc; ++k) m = fmaxf(m, (float)base[(size_t)k * N + c]);
        out[(size_t)q * N + c] = m;
    }
}

// ---------------------------------------------------------------------------
// FC head helpers (M padded to 128 rows for the 128x128 WMMA tiler).
// ---------------------------------------------------------------------------
__global__ void pack_fc(const float* __restrict__ x, _Float16* __restrict__ A,
                        int Mreal, int K)
{
    const int r = blockIdx.x;
    for (int c = threadIdx.x; c < K; c += blockDim.x)
        A[(size_t)r * K + c] = (_Float16)(r < Mreal ? x[(size_t)r * K + c] : 0.f);
}

__global__ void fc_bn(_Float16* __restrict__ H, const float* __restrict__ g,
                      const float* __restrict__ be, int Mreal, int N)
{
    const int c = blockIdx.x * blockDim.x + threadIdx.x;
    if (c >= N) return;
    float s = 0.f, s2 = 0.f;
    for (int r = 0; r < Mreal; ++r) {
        const float v = (float)H[(size_t)r * N + c];
        s += v; s2 += v * v;
    }
    const float mean = s / Mreal;
    const float rs = rsqrtf(s2 / Mreal - mean * mean + 1e-5f);
    for (int r = 0; r < Mreal; ++r) {
        const float v = ((float)H[(size_t)r * N + c] - mean) * rs * g[c] + be[c];
        H[(size_t)r * N + c] = (_Float16)relu_(v);
    }
}

__global__ __launch_bounds__(256)
void fc3_softmax(const _Float16* __restrict__ H, const float* __restrict__ w3,
                 const float* __restrict__ b3, float* __restrict__ out)
{
    __shared__ float logits[8];
    const int t = threadIdx.x, r = t >> 5, lane = t & 31;
    float s = 0.f;
    for (int c = lane; c < 256; c += 32) s += (float)H[r * 256 + c] * w3[c];
    for (int off = 16; off > 0; off >>= 1) s += __shfl_down(s, off, 32);
    if (lane == 0) logits[r] = s + b3[0];
    __syncthreads();
    if (t < 2) {
        float m = -1e30f;
        for (int j = 0; j < 4; ++j) m = fmaxf(m, logits[t * 4 + j]);
        float e[4], den = 0.f;
        for (int j = 0; j < 4; ++j) { e[j] = __expf(logits[t * 4 + j] - m); den += e[j]; }
        for (int j = 0; j < 4; ++j) {
            out[t * 4 + j]     = logits[t * 4 + j];   // logits
            out[8 + t * 4 + j] = e[j] / den;          // softmax
        }
    }
}

// ---------------------------------------------------------------------------
// Host orchestration
// ---------------------------------------------------------------------------
static inline int pad32(int x) { return (x + 31) & ~31; }

extern "C" void kernel_launch(void* const* d_in, const int* in_sizes, int n_in,
                              void* d_out, int out_size, void* d_ws, size_t ws_size,
                              hipStream_t stream)
{
    (void)in_sizes; (void)n_in; (void)out_size; (void)ws_size;

    const float* desc = (const float*)d_in[0];   // (8, 2048, 256)
    const float* kp   = (const float*)d_in[1];   // (8, 2048, 3)
    int pi = 2;
    const float *saw[11], *sag[11], *sabe[11];
    for (int l = 0; l < 11; ++l) {
        saw[l]  = (const float*)d_in[pi++];
        pi++;                                   // bias (cancels under BN)
        sag[l]  = (const float*)d_in[pi++];
        sabe[l] = (const float*)d_in[pi++];
    }
    const float* fc1_w = (const float*)d_in[pi++]; pi++;  // fc1_b cancels
    const float* bn1_g = (const float*)d_in[pi++];
    const float* bn1_b = (const float*)d_in[pi++];
    const float* fc2_w = (const float*)d_in[pi++]; pi++;  // fc2_b cancels
    const float* bn2_g = (const float*)d_in[pi++];
    const float* bn2_b = (const float*)d_in[pi++];
    const float* fc3_w = (const float*)d_in[pi++];
    const float* fc3_b = (const float*)d_in[pi++];

    // Workspace carve-up
    char* ws = (char*)d_ws;
    size_t off = 0;
    auto carve = [&](size_t bytes) -> void* {
        void* p = ws + off; off = (off + bytes + 255) & ~(size_t)255; return p;
    };
    _Float16* H0    = (_Float16*)carve(140ull << 20);   // ping (max 65536x1024 f16)
    _Float16* H1    = (_Float16*)carve(140ull << 20);   // pong
    _Float16* Wh    = (_Float16*)carve(10ull << 20);    // f16 weights (max 2048x2048)
    float*    stats = (float*)carve(2 * STATS_STRIDE * sizeof(float));
    float*    xyzA  = (float*)carve((size_t)BQ * 2048 * 3 * 4);
    float*    xyzB  = (float*)carve((size_t)BQ * 2048 * 3 * 4);
    float*    ptsA  = (float*)carve(17ull << 20);       // pooled feats (max 16 MB)
    float*    ptsB  = (float*)carve(17ull << 20);
    int*      fidx  = (int*)carve((size_t)BQ * 1024 * 4);
    int*      bidx  = (int*)carve((size_t)BQ * 1024 * 32 * 4);

    struct SACfg { int S, Kc; float r; int mlp[3]; int nmlp; int Cin; };
    const SACfg cfg[4] = {
        {1024,  8, 0.2f, {256,  512,  512}, 3,  259},
        { 512, 16, 0.4f, {512,  512, 1024}, 3,  515},
        { 128, 32, 0.8f, {1024, 1024,   0}, 2, 1027},
        {   0,  0, 0.0f, {1024, 2048, 2048}, 3, 1027},   // group_all
    };

    const float* curXyz = kp;
    const float* curPts = desc;
    int curN = 2048, curC = 256;
    float* xyzBufs[2] = {xyzA, xyzB};
    float* ptsBufs[2] = {ptsA, ptsB};
    int xb = 0, pb = 0;
    int layerIdx = 0;
    float* xfc = nullptr;

    for (int st = 0; st < 4; ++st) {
        int M, Kpad;
        if (st < 3) {
            const int S = cfg[st].S, Kc = cfg[st].Kc, Q = BQ * S;
            float* nxyz = xyzBufs[xb]; xb ^= 1;
            fps_kernel<<<BQ, 256, 0, stream>>>(curXyz, curN, S, fidx);
            gather_newxyz<<<(Q + 255) / 256, 256, 0, stream>>>(curXyz, fidx, nxyz, curN, S, Q);
            ballquery_kernel<<<(Q + 7) / 8, 256, 0, stream>>>(
                curXyz, nxyz, cfg[st].r * cfg[st].r, curN, S, Kc, bidx, Q);
            M = Q * Kc;
            Kpad = pad32(3 + curC);
            group_kernel<<<M, 256, 0, stream>>>(curXyz, nxyz, curPts, bidx, H0,
                                                curN, S, Kc, curC, Kpad);
            curXyz = nxyz;
        } else {
            M = BQ * curN;               // 8 * 128 = 1024 rows
            Kpad = pad32(3 + curC);
            group_all_kernel<<<M, 256, 0, stream>>>(curXyz, curPts, H0, curN, curC, Kpad);
        }

        // MLP stack: GEMM -> BN stats -> BN+ReLU (in place, re-emits f16)
        _Float16* Ain = H0; _Float16* Aout = H1;
        for (int j = 0; j < cfg[st].nmlp; ++j) {
            const int Cin  = (j == 0) ? cfg[st].Cin : cfg[st].mlp[j - 1];
            const int Cout = cfg[st].mlp[j];
            convert_w<<<Cout, 256, 0, stream>>>(saw[layerIdx], Wh, Cout, Cin, Kpad);
            dim3 gg(Cout / 128, M / 128);
            gemm_f16_wmma<<<gg, 256, 0, stream>>>(Ain, Wh, Aout, M, Cout, Kpad);
            hipMemsetAsync(stats, 0, 2 * STATS_STRIDE * sizeof(float), stream);
            dim3 gs((Cout + 63) / 64, (M + 2047) / 2048);
            bn_stats<<<gs, 256, 0, stream>>>(Aout, stats, M, Cout, 2048);
            const size_t tot = (size_t)M * Cout;
            bn_apply<<<(unsigned)((tot + 255) / 256), 256, 0, stream>>>(
                Aout, stats, sag[layerIdx], sabe[layerIdx], M, Cout);
            _Float16* tmp = Ain; Ain = Aout; Aout = tmp;
            Kpad = Cout;                 // all Couts are multiples of 32
            ++layerIdx;
        }
        const int lastC = cfg[st].mlp[cfg[st].nmlp - 1];

        if (st < 3) {
            float* pooled = ptsBufs[pb]; pb ^= 1;
            maxpool_kernel<<<BQ * cfg[st].S, 256, 0, stream>>>(Ain, pooled, cfg[st].Kc, lastC);
            curPts = pooled; curN = cfg[st].S; curC = lastC;
        } else {
            xfc = ptsBufs[pb];           // (8, 2048) global features
            maxpool_kernel<<<BQ, 256, 0, stream>>>(Ain, xfc, curN, lastC);
        }
    }

    // ---- FC head (M padded 8 -> 128 for the 128x128 WMMA tiler) ----
    pack_fc<<<128, 256, 0, stream>>>(xfc, H0, BQ, 2048);
    convert_w<<<512, 256, 0, stream>>>(fc1_w, Wh, 512, 2048, 2048);
    gemm_f16_wmma<<<dim3(512 / 128, 1), 256, 0, stream>>>(H0, Wh, H1, 128, 512, 2048);
    fc_bn<<<2, 256, 0, stream>>>(H1, bn1_g, bn1_b, BQ, 512);

    convert_w<<<256, 256, 0, stream>>>(fc2_w, Wh, 256, 512, 512);
    gemm_f16_wmma<<<dim3(256 / 128, 1), 256, 0, stream>>>(H1, Wh, H0, 128, 256, 512);
    fc_bn<<<1, 256, 0, stream>>>(H0, bn2_g, bn2_b, BQ, 256);

    fc3_softmax<<<1, 256, 0, stream>>>(H0, fc3_w, fc3_b, (float*)d_out);
}
